// SelfAttention_88570815578531
// MI455X (gfx1250) — matrix operand order repaired and checked by reading, not compiled, not run on hardware
//
#include <hip/hip_runtime.h>

typedef _Float16 f16;
typedef __attribute__((ext_vector_type(16))) _Float16 v16h;
typedef __attribute__((ext_vector_type(8)))  float    v8f;
typedef __attribute__((ext_vector_type(4)))  unsigned int u32x4;
typedef __attribute__((ext_vector_type(8)))  int i32x8;
typedef __attribute__((ext_vector_type(4)))  int i32x4;

#define B_  8
#define C_  256
#define H_  64
#define W_  64
#define HW  4096
#define CQ  32
#define CV  128
#define M_  1024
#define OCP 192

union F16x16 { v16h v; u32x4 q[2]; };
union F16x8  { f16 h[8]; u32x4 q; };

// A-fragment K index map (16-bit A 16x32, ISA 7.12.2):
// lanes 0-15: K = {0..7, 16..23}; lanes 16-31: K = {8..15, 24..31}
__device__ __forceinline__ int amap(int half, int i) {
    return (i < 8) ? (half * 8 + i) : (8 + half * 8 + i);
}

// ---------------------------------------------------------------------------
// TDM 2-D tile load: global -> LDS.  Descriptor per CDNA5 ISA
// §8.3 (group0: count/lds_addr/global_addr/type=2) and §8.4 (group1: data_size,
// tensor dims, tile dims, 48-bit dim0 stride).  Remaining groups zero (2-D).
// 6-arg builtin on this toolchain: (u32x4, i32x8, i32x4, i32x4, i32x8, cpol).
// ---------------------------------------------------------------------------
__device__ __forceinline__ void tdm_load_2d(unsigned lds_off, const void* gaddr,
                                            unsigned tensor_d0, unsigned tensor_d1,
                                            unsigned tile_d0, unsigned tile_d1,
                                            unsigned stride0, unsigned ds_code) {
    unsigned long long ga = (unsigned long long)gaddr;
    u32x4 g0;
    g0[0] = 1u;                                            // count=1 (valid), user mode
    g0[1] = lds_off;                                       // lds_addr [63:32]
    g0[2] = (unsigned)(ga & 0xffffffffull);                // global_addr low
    g0[3] = (unsigned)((ga >> 32) & 0x1ffffffull)          // global_addr [56:32]
          | (2u << 30);                                    // type=2 ("image")
    i32x8 g1;
    g1[0] = (int)(ds_code << 16);                          // wg_mask=0, data_size
    g1[1] = (int)(tensor_d0 << 16);                        // tensor_dim0[15:0] @ bits 63:48
    g1[2] = (int)((tensor_d0 >> 16) & 0xffffu) | (int)(tensor_d1 << 16);
    g1[3] = (int)((tensor_d1 >> 16) & 0xffffu) | (int)(tile_d0 << 16);
    g1[4] = (int)(tile_d1 & 0xffffu);                      // tile_dim1; tile_dim2=0
    g1[5] = (int)stride0;                                  // dim0 stride low32
    g1[6] = 0;                                             // stride0 hi16 / stride1 lo16
    g1[7] = 0;
    i32x4 z4 = {0, 0, 0, 0};
    i32x8 z8 = {0, 0, 0, 0, 0, 0, 0, 0};
    __builtin_amdgcn_tensor_load_to_lds(g0, g1, z4, z4, z8, 0);
}

// ---------------------------------------------------------------------------
// Kernel 1: fused q/k/v 1x1-conv projection.  One block = 12 waves covers a
// 16-column x-slab for ALL 192 output channels; x tile staged once via TDM.
// GEMM orientation: A = W rows (M=oc), B = x tile (N=n) -> contiguous f16 store.
// ---------------------------------------------------------------------------
__global__ __launch_bounds__(384)
void k_qkv_proj(const float* __restrict__ x, const float* __restrict__ wq,
                const float* __restrict__ wk, const float* __restrict__ wv,
                f16* __restrict__ proj) {
    __shared__ float xs[C_ * 16];                 // [K=256][n=16] f32, 16KB
    const int tid  = threadIdx.x;
    const int lane = tid & 31;
    const int wid  = tid >> 5;                    // wave id 0..11 -> oc tile
    const int half = lane >> 4, l16 = lane & 15;
    const int b  = blockIdx.y;
    const int n0 = blockIdx.x * 16;
    const float* xb = x + (size_t)b * C_ * HW;

    if (wid == 0) {
        tdm_load_2d((unsigned)(size_t)&xs[0], xb + n0,
                    /*tensor_d0=*/HW, /*tensor_d1=*/C_,
                    /*tile_d0=*/16,  /*tile_d1=*/C_,
                    /*stride0=*/HW,  /*ds_code=*/2 /*4B*/);
        __builtin_amdgcn_s_wait_tensorcnt(0);
    }
    __syncthreads();

    const int oc0 = wid * 16;
    const float* wbase; int ocl;
    if (oc0 < 32)      { wbase = wq; ocl = oc0; }
    else if (oc0 < 64) { wbase = wk; ocl = oc0 - 32; }
    else               { wbase = wv; ocl = oc0 - 64; }

    v8f acc = {};
    #pragma unroll
    for (int kc = 0; kc < 8; ++kc) {
        const int K0 = kc * 32;
        v16h a, bf;
        const float* wr = wbase + (size_t)(ocl + l16) * C_ + K0;
        #pragma unroll
        for (int i = 0; i < 16; ++i) a[i] = (f16)wr[amap(half, i)];
        #pragma unroll
        for (int i = 0; i < 16; ++i) bf[i] = (f16)xs[(K0 + amap(half, i)) * 16 + l16];
        acc = __builtin_amdgcn_wmma_f32_16x16x32_f16(false, a, false, bf,
                                                     (short)0, acc, false, false);
    }
    // D[M=oc][N=n]: lane has fixed n = n0+l16, 8 consecutive oc -> one b128 store
    F16x8 st;
    #pragma unroll
    for (int v = 0; v < 8; ++v) st.h[v] = (f16)acc[v];
    f16* pr = proj + (size_t)b * HW * OCP + (size_t)(n0 + l16) * OCP + oc0 + 8 * half;
    *(u32x4*)pr = st.q;
}

// ---------------------------------------------------------------------------
// Kernel 2: 2x2 max-pool -> kT[b][m][32] (m-major), vP[b][cv][m] (cv-major,
// so attention V chunks are perfect 2-D TDM tiles).
// ---------------------------------------------------------------------------
__global__ __launch_bounds__(256)
void k_pool(const f16* __restrict__ proj, f16* __restrict__ kT, f16* __restrict__ vP) {
    int idx = blockIdx.x * 256 + threadIdx.x;
    const int total = B_ * M_ * 160;
    if (idx >= total) return;
    int ch = idx % 160;
    int m  = (idx / 160) % M_;
    int b  = idx / (160 * M_);
    int mh = m >> 5, mw = m & 31;
    int n00 = (mh * 2) * W_ + mw * 2;
    const f16* pb = proj + (size_t)b * HW * OCP;
    int oc = 32 + ch;
    float v0 = (float)pb[(size_t)n00 * OCP + oc];
    float v1 = (float)pb[(size_t)(n00 + 1) * OCP + oc];
    float v2 = (float)pb[(size_t)(n00 + W_) * OCP + oc];
    float v3 = (float)pb[(size_t)(n00 + W_ + 1) * OCP + oc];
    float r = fmaxf(fmaxf(v0, v1), fmaxf(v2, v3));
    if (ch < 32) kT[((size_t)b * M_ + m) * CQ + ch] = (f16)r;
    else         vP[((size_t)b * CV + (ch - 32)) * M_ + m] = (f16)r;
}

// ---------------------------------------------------------------------------
// Kernel 3: fused flash attention.  4 waves per workgroup (4 query tiles, 64
// rows) SHARE one TDM double-buffered V chunk (32 m x 128 cv, stored [cv][32]
// so every V B-fragment is a contiguous 32B LDS read).  Wave 0 drives the DMA;
// two workgroup barriers per chunk publish/recycle the buffers.  Final ctx
// tile is transposed through per-wave LDS scratch into coalesced b128 stores.
// ---------------------------------------------------------------------------
__global__ __launch_bounds__(128)
void k_attn(const f16* __restrict__ proj, const f16* __restrict__ kT,
            const f16* __restrict__ vP, f16* __restrict__ ctx) {
    __shared__ f16 Vlds[2][CV * 32];              // 16KB, shared by all 4 waves
    __shared__ f16 Sbuf[4][16 * CV];              // 4KB per wave: P buf + ctx staging
    const int tid  = threadIdx.x;
    const int wid  = tid >> 5;                    // wave 0..3 -> query tile
    const int lane = tid & 31;
    const int half = lane >> 4, l16 = lane & 15;
    const int b  = blockIdx.z;
    const int n0 = (blockIdx.x * 4 + wid) * 16;
    f16* Pb = &Sbuf[wid][0];                      // P round-trip region (16x32)

    // Persistent Q A-fragment: two aligned 16B global loads
    const f16* prow = proj + (size_t)b * HW * OCP + (size_t)(n0 + l16) * OCP;
    F16x16 qa;
    qa.q[0] = *(const u32x4*)(prow + half * 16);
    qa.q[1] = *(const u32x4*)(prow + half * 16 + 8);

    const f16* kTb = kT + (size_t)b * M_ * CQ;
    const f16* vPb = vP + (size_t)b * CV * M_;

    float m_run[8], l_run[8];
    #pragma unroll
    for (int v = 0; v < 8; ++v) { m_run[v] = -3.0e38f; l_run[v] = 0.f; }
    v8f zero = {};
    v8f cacc[8];
    #pragma unroll
    for (int t = 0; t < 8; ++t) cacc[t] = zero;

    // Prologue: wave 0 starts DMA of V chunk 0
    if (wid == 0)
        tdm_load_2d((unsigned)(size_t)&Vlds[0][0], vPb,
                    /*tensor_d0=*/M_, /*tensor_d1=*/CV,
                    /*tile_d0=*/32,  /*tile_d1=*/CV,
                    /*stride0=*/M_,  /*ds_code=*/1 /*2B*/);

    for (int mc = 0; mc < 32; ++mc) {
        const int m0 = mc * 32;
        const int buf = mc & 1;
        __syncthreads();                          // (A) buf^1 free to overwrite
        if (wid == 0) {
            if (mc + 1 < 32) {
                tdm_load_2d((unsigned)(size_t)&Vlds[buf ^ 1][0], vPb + (m0 + 32),
                            M_, CV, 32, CV, M_, 1);
                __builtin_amdgcn_s_wait_tensorcnt(1);   // chunk mc complete
            } else {
                __builtin_amdgcn_s_wait_tensorcnt(0);
            }
        }
        __syncthreads();                          // (B) chunk mc visible to all
        asm volatile("" ::: "memory");

        // K B-fragments (contiguous 32B global loads)
        F16x16 kf0, kf1;
        {
            const f16* kr0 = kTb + (size_t)(m0 + l16) * CQ + half * 16;
            const f16* kr1 = kTb + (size_t)(m0 + 16 + l16) * CQ + half * 16;
            kf0.q[0] = *(const u32x4*)(kr0);
            kf0.q[1] = *(const u32x4*)(kr0 + 8);
            kf1.q[0] = *(const u32x4*)(kr1);
            kf1.q[1] = *(const u32x4*)(kr1 + 8);
        }
        v8f s0 = __builtin_amdgcn_wmma_f32_16x16x32_f16(false, qa.v, false, kf0.v,
                                                        (short)0, zero, false, false);
        v8f s1 = __builtin_amdgcn_wmma_f32_16x16x32_f16(false, qa.v, false, kf1.v,
                                                        (short)0, zero, false, false);
        // Online softmax: row r's 16 m-values live in one 16-lane half
        float scale[8], p0[8], p1[8];
        #pragma unroll
        for (int v = 0; v < 8; ++v) {
            float r = fmaxf(s0[v], s1[v]);
            r = fmaxf(r, __shfl_xor(r, 1, 32));
            r = fmaxf(r, __shfl_xor(r, 2, 32));
            r = fmaxf(r, __shfl_xor(r, 4, 32));
            r = fmaxf(r, __shfl_xor(r, 8, 32));
            float mn = fmaxf(m_run[v], r);
            scale[v] = __expf(m_run[v] - mn);
            m_run[v] = mn;
            p0[v] = __expf(s0[v] - mn);
            p1[v] = __expf(s1[v] - mn);
            float rs = p0[v] + p1[v];
            rs += __shfl_xor(rs, 1, 32);
            rs += __shfl_xor(rs, 2, 32);
            rs += __shfl_xor(rs, 4, 32);
            rs += __shfl_xor(rs, 8, 32);
            l_run[v] = l_run[v] * scale[v] + rs;
        }
        #pragma unroll
        for (int t = 0; t < 8; ++t)
            #pragma unroll
            for (int v = 0; v < 8; ++v) cacc[t][v] *= scale[v];

        // P: C/D layout -> LDS [16 n][32 m] -> reload as A-fragment (2x b128)
        #pragma unroll
        for (int v = 0; v < 8; ++v) {
            Pb[(v + 8 * half) * 32 + l16]      = (f16)p0[v];
            Pb[(v + 8 * half) * 32 + 16 + l16] = (f16)p1[v];
        }
        asm volatile("s_wait_dscnt 0" ::: "memory");
        F16x16 pa;
        pa.q[0] = *(const u32x4*)(&Pb[l16 * 32 + half * 16]);
        pa.q[1] = *(const u32x4*)(&Pb[l16 * 32 + half * 16 + 8]);

        // ctx += P(16x32) @ V-chunk(32x128); V B-fragments: contiguous LDS b128
        #pragma unroll
        for (int t = 0; t < 8; ++t) {
            F16x16 vf;
            const f16* vr = &Vlds[buf][(t * 16 + l16) * 32 + half * 16];
            vf.q[0] = *(const u32x4*)(vr);
            vf.q[1] = *(const u32x4*)(vr + 8);
            cacc[t] = __builtin_amdgcn_wmma_f32_16x16x32_f16(false, pa.v, false, vf.v,
                                                             (short)0, cacc[t], false, false);
        }
    }

    // Normalize, transpose through per-wave LDS, coalesced b128 stores.
    float inv[8];
    #pragma unroll
    for (int v = 0; v < 8; ++v) inv[v] = 1.0f / l_run[v];
    #pragma unroll
    for (int t = 0; t < 8; ++t)
        #pragma unroll
        for (int v = 0; v < 8; ++v)
            Sbuf[wid][(v + 8 * half) * CV + t * 16 + l16] = (f16)(cacc[t][v] * inv[v]);
    asm volatile("s_wait_dscnt 0" ::: "memory");

    f16* cb = ctx + (size_t)b * HW * CV;
    #pragma unroll
    for (int j = 0; j < 8; ++j) {                 // 2048 f16 = 256 x u32x4
        int u   = j * 32 + lane;                  // u32x4 index
        int row = u >> 4;                         // 16 u32x4 per 128-f16 row
        int col = (u & 15) * 8;                   // f16 offset within row
        u32x4 d = *(const u32x4*)(&Sbuf[wid][row * CV + col]);
        *(u32x4*)(cb + (size_t)(n0 + row) * CV + col) = d;
    }
}

// ---------------------------------------------------------------------------
// Kernel 4: out = gamma * (wo @ ctx) + x.  D[M=C][N=n], K = 128 in 4 chunks.
// ---------------------------------------------------------------------------
__global__ __launch_bounds__(32)
void k_out(const float* __restrict__ x, const float* __restrict__ wo,
           const f16* __restrict__ ctx, const float* __restrict__ gamma,
           float* __restrict__ out) {
    const int lane = threadIdx.x & 31;
    const int half = lane >> 4, l16 = lane & 15;
    const int b  = blockIdx.z;
    const int n0 = blockIdx.x * 16;
    const int c0 = blockIdx.y * 16;
    const f16* cb = ctx + (size_t)b * HW * CV + (size_t)(n0 + l16) * CV;

    v8f acc = {};
    #pragma unroll
    for (int kc = 0; kc < 4; ++kc) {
        const int K0 = kc * 32;
        v16h a;
        F16x16 bf;
        const float* wr = wo + (size_t)(c0 + l16) * CV + K0;
        #pragma unroll
        for (int i = 0; i < 16; ++i) a[i] = (f16)wr[amap(half, i)];
        bf.q[0] = *(const u32x4*)(cb + K0 + half * 8);
        bf.q[1] = *(const u32x4*)(cb + K0 + 16 + half * 8);
        acc = __builtin_amdgcn_wmma_f32_16x16x32_f16(false, a, false, bf.v,
                                                     (short)0, acc, false, false);
    }
    const float g = gamma[0];
    const float* xb = x + (size_t)b * C_ * HW;
    float* ob = out + (size_t)b * C_ * HW;
    #pragma unroll
    for (int v = 0; v < 8; ++v) {
        int c = c0 + v + 8 * half;
        size_t idx = (size_t)c * HW + n0 + l16;
        ob[idx] = g * acc[v] + xb[idx];          // coalesced across 16 lanes
    }
}

// ---------------------------------------------------------------------------
extern "C" void kernel_launch(void* const* d_in, const int* in_sizes, int n_in,
                              void* d_out, int out_size, void* d_ws, size_t ws_size,
                              hipStream_t stream) {
    const float* x     = (const float*)d_in[0];
    const float* wq    = (const float*)d_in[1];
    const float* wk    = (const float*)d_in[2];
    const float* wv    = (const float*)d_in[3];
    const float* wo    = (const float*)d_in[4];
    const float* gamma = (const float*)d_in[5];
    float* out = (float*)d_out;

    f16* ws   = (f16*)d_ws;
    f16* proj = ws;                               // B*HW*192   f16 (12.6 MB)
    f16* kT   = proj + (size_t)B_ * HW * OCP;     // B*1024*32  f16 (0.5 MB)
    f16* vP   = kT   + (size_t)B_ * M_ * CQ;      // B*128*1024 f16 (2 MB)
    f16* ctx  = vP   + (size_t)B_ * CV * M_;      // B*HW*128   f16 (8 MB)

    k_qkv_proj<<<dim3(HW / 16, B_), 384, 0, stream>>>(x, wq, wk, wv, proj);

    const int total = B_ * M_ * 160;
    k_pool<<<(total + 255) / 256, 256, 0, stream>>>(proj, kT, vP);

    k_attn<<<dim3(HW / 64, 1, B_), 128, 0, stream>>>(proj, kT, vP, ctx);

    k_out<<<dim3(HW / 16, C_ / 16, B_), 32, 0, stream>>>(x, wo, ctx, gamma, out);
}
